// SelfAttention_8392366097086
// MI455X (gfx1250) — matrix operand order repaired and checked by reading, not compiled, not run on hardware
//
#include <hip/hip_runtime.h>
#include <hip/hip_bf16.h>

typedef __attribute__((ext_vector_type(16))) __bf16 v16bf;
typedef __attribute__((ext_vector_type(8)))  float  v8f;

union Frag16 { v16bf v; unsigned u[8]; uint4 q2[2]; };

__device__ inline v8f wmma_bf16(const Frag16& a, const Frag16& b, v8f c) {
    return __builtin_amdgcn_wmma_f32_16x16x32_bf16(
        false, a.v, false, b.v, (short)0, c, false, false);
}

__device__ inline unsigned pack_bf16(float a, float b) {
    union { __bf16 h[2]; unsigned u; } p;
    p.h[0] = (__bf16)a; p.h[1] = (__bf16)b;
    return p.u;
}

// A fragment (16x32 bf16): per lane two contiguous 16B runs -> 2x ds_load_b128.
__device__ inline void load_a_frag(const __bf16* base, int ld, int lm, int lh, Frag16& f) {
    const __bf16* p = base + lm * ld + lh * 8;
    f.q2[0] = *(const uint4*)p;          // K = lh*8 .. lh*8+7
    f.q2[1] = *(const uint4*)(p + 16);   // K = 16+lh*8 .. 16+lh*8+7
}

// B fragment (32x16, tile stored transposed Bt[n][k]): two 16B runs, K = lh*8.. and 16+lh*8...
__device__ inline void load_b_frag(const __bf16* base, int ld, int lm, int lh, Frag16& f) {
    const __bf16* p = base + lm * ld + lh * 8;
    f.q2[0] = *(const uint4*)p;
    f.q2[1] = *(const uint4*)(p + 16);
}

// width-16 xor-shuffle max reduction via v_permlane16_b32 (pure VALU, no DS)
__device__ inline float permlane16_f(float x, unsigned s0, unsigned s1) {
    unsigned u = __builtin_bit_cast(unsigned, x);
    u = __builtin_amdgcn_permlane16(u, u, s0, s1, false, false);
    return __builtin_bit_cast(float, u);
}
__device__ inline float rowmax16(float x) {
    x = fmaxf(x, permlane16_f(x, 0x67452301u, 0xEFCDAB89u)); // xor 1
    x = fmaxf(x, permlane16_f(x, 0x45670123u, 0xCDEF89ABu)); // xor 3
    x = fmaxf(x, permlane16_f(x, 0x01234567u, 0x89ABCDEFu)); // xor 7
    x = fmaxf(x, permlane16_f(x, 0xFEDCBA98u, 0x76543210u)); // xor 8
    return x;
}

// ---------------------------------------------------------------------------
// GEMM: C[M,N] = (A[M,K] @ W[K,N] + bias) * scale   (A fp32|bf16; C bf16|fp32)
// 256 thr = 8 waves; macro tile 128x128; wave tile 32(M) x 64(N); K step 32.
// ---------------------------------------------------------------------------
#define GT_M 128
#define GT_N 128
#define GT_K 32
#define GLD  40   // LDS row stride (elements); %8==0 -> 16B aligned rows

template <bool A_IS_BF16, bool OUT_BF16>
__global__ __launch_bounds__(256) void gemm_bias_kernel(
    const void* __restrict__ Ap, const float* __restrict__ W,
    const float* __restrict__ bias, void* __restrict__ Cp,
    int M, int N, int K, float scale)
{
    __shared__ alignas(16) __bf16 As[GT_M * GLD];
    __shared__ alignas(16) __bf16 Ws[GT_N * GLD];

    const int t = threadIdx.x;
    const int wave = t >> 5, lane = t & 31;
    const int lm = lane & 15, lh = lane >> 4;
    const int wm = wave & 3;   // M sub-block
    const int wn = wave >> 2;  // N sub-block
    const int m0 = blockIdx.y * GT_M;
    const int n0 = blockIdx.x * GT_N;

    v8f acc[2][4];
#pragma unroll
    for (int i = 0; i < 2; ++i)
#pragma unroll
        for (int j = 0; j < 4; ++j)
#pragma unroll
            for (int r = 0; r < 8; ++r) acc[i][j][r] = 0.0f;

    for (int k0 = 0; k0 < K; k0 += GT_K) {
        // ---- stage A tile (128x32) ----
        if (A_IS_BF16) {
#pragma unroll
            for (int i = 0; i < 2; ++i) {
                int e = t + i * 256;              // 0..511 (8-elem groups)
                int m = e >> 2, k8 = (e & 3) * 8;
                *(uint4*)&As[m * GLD + k8] =
                    *(const uint4*)((const __bf16*)Ap + (size_t)(m0 + m) * K + k0 + k8);
            }
        } else {
#pragma unroll
            for (int i = 0; i < 4; ++i) {
                int e = t + i * 256;              // 0..1023 (4-elem groups)
                int m = e >> 3, k4 = (e & 7) * 4;
                float4 fa = *(const float4*)((const float*)Ap +
                                             (size_t)(m0 + m) * K + k0 + k4);
                uint2 pk;
                pk.x = pack_bf16(fa.x, fa.y);
                pk.y = pack_bf16(fa.z, fa.w);
                *(uint2*)&As[m * GLD + k4] = pk;
            }
        }
        // ---- stage W tile transposed: Ws[n][k] = W[k0+k][n0+n] ----
#pragma unroll
        for (int i = 0; i < 4; ++i) {
            int e = t + i * 256;                  // 0..1023
            int kk = e >> 5, n4 = (e & 31) * 4;
            float4 fw = *(const float4*)(W + (size_t)(k0 + kk) * N + n0 + n4);
            Ws[(n4 + 0) * GLD + kk] = (__bf16)fw.x;
            Ws[(n4 + 1) * GLD + kk] = (__bf16)fw.y;
            Ws[(n4 + 2) * GLD + kk] = (__bf16)fw.z;
            Ws[(n4 + 3) * GLD + kk] = (__bf16)fw.w;
        }
        // ---- prefetch next K tiles into L2 while this tile computes ----
        if (k0 + GT_K < K) {
            int pr = t >> 1, ph = (t & 1) * 16;
            if (A_IS_BF16)
                __builtin_prefetch((const __bf16*)Ap + (size_t)(m0 + pr) * K + k0 + GT_K + ph, 0, 1);
            else
                __builtin_prefetch((const float*)Ap + (size_t)(m0 + pr) * K + k0 + GT_K + ph, 0, 1);
            int wr = t >> 3, wc = (t & 7) * 16;
            __builtin_prefetch(W + (size_t)(k0 + GT_K + wr) * N + n0 + wc, 0, 1);
        }
        __syncthreads();

        Frag16 af[2], bf[4];
#pragma unroll
        for (int i = 0; i < 2; ++i)
            load_a_frag(&As[(wm * 32 + i * 16) * GLD], GLD, lm, lh, af[i]);
#pragma unroll
        for (int j = 0; j < 4; ++j)
            load_b_frag(&Ws[(wn * 64 + j * 16) * GLD], GLD, lm, lh, bf[j]);
#pragma unroll
        for (int i = 0; i < 2; ++i)
#pragma unroll
            for (int j = 0; j < 4; ++j)
                acc[i][j] = wmma_bf16(af[i], bf[j], acc[i][j]);
        __syncthreads();
    }

#pragma unroll
    for (int i = 0; i < 2; ++i)
#pragma unroll
        for (int j = 0; j < 4; ++j)
#pragma unroll
            for (int r = 0; r < 8; ++r) {
                int row = m0 + wm * 32 + i * 16 + r + 8 * lh;
                int col = n0 + wn * 64 + j * 16 + lm;
                float vv = (acc[i][j][r] + bias[col]) * scale;
                if (OUT_BF16)
                    ((__bf16*)Cp)[(size_t)row * N + col] = (__bf16)vv;
                else
                    ((float*)Cp)[(size_t)row * N + col] = vv;
            }
}

// ---------------------------------------------------------------------------
// Flash attention. q pre-scaled by 1/sqrt(Hd). bf16 [B,S,D] in/out.
// 256 thr = 8 waves; 128 queries/block; 64-key chunks; l via ones-column WMMA.
// K tile staged with global_load_async_to_lds_b128 (ASYNCcnt path).
// ---------------------------------------------------------------------------
#define ATT_S  2048
#define ATT_D  1024
#define ATT_HD 64
#define ATT_KC 64
#define LDK 72
#define LDV 72
#define LDP 72

__global__ __launch_bounds__(256) void attn_kernel(
    const __bf16* __restrict__ q, const __bf16* __restrict__ k,
    const __bf16* __restrict__ v, __bf16* __restrict__ o)
{
    __shared__ alignas(16) __bf16 Kt[ATT_KC * LDK];      // [key][d]
    __shared__ alignas(16) __bf16 Vt[ATT_HD * LDV];      // [d][key]
    __shared__ alignas(16) __bf16 Ps[8 * 16 * LDP];      // per-wave P 16x64

    const int t = threadIdx.x;
    const int wave = t >> 5, lane = t & 31;
    const int lm = lane & 15, lh = lane >> 4;
    const int b = blockIdx.z, h = blockIdx.y;
    const int q0 = blockIdx.x * 128;
    const size_t headoff = (size_t)h * ATT_HD;

    // Q fragments for this wave's 16 rows (two K=32 halves of Hd=64)
    Frag16 qa[2];
    {
        const __bf16* qrow =
            q + ((size_t)b * ATT_S + q0 + wave * 16 + lm) * ATT_D + headoff;
#pragma unroll
        for (int f = 0; f < 2; ++f) {
            const __bf16* p = qrow + f * 32 + lh * 8;
            qa[f].q2[0] = *(const uint4*)p;
            qa[f].q2[1] = *(const uint4*)(p + 16);
        }
    }

    Frag16 onesf;   // bf16 1.0 everywhere: rowsum accumulator B operand
#pragma unroll
    for (int i = 0; i < 8; ++i) onesf.u[i] = 0x3f803f80u;

    v8f oacc[4], osum;
#pragma unroll
    for (int j = 0; j < 4; ++j)
#pragma unroll
        for (int r = 0; r < 8; ++r) oacc[j][r] = 0.0f;
#pragma unroll
    for (int r = 0; r < 8; ++r) osum[r] = 0.0f;
    float mrow[8];
#pragma unroll
    for (int r = 0; r < 8; ++r) mrow[r] = -3.0e38f;

    for (int kc = 0; kc < ATT_S; kc += ATT_KC) {
        __syncthreads();
        const __bf16* kbase = k + ((size_t)b * ATT_S + kc) * ATT_D + headoff;
        const __bf16* vbase = v + ((size_t)b * ATT_S + kc) * ATT_D + headoff;

        // stage K chunk: async DMA global -> LDS, 16B per lane per issue
#pragma unroll
        for (int i = 0; i < 2; ++i) {
            int e = t + i * 256;                 // 0..511
            int key = e >> 3, d8 = (e & 7) * 8;
            unsigned lds = (unsigned)(uintptr_t)&Kt[key * LDK + d8];
            unsigned long long ga =
                (unsigned long long)(uintptr_t)(kbase + (size_t)key * ATT_D + d8);
            asm volatile("global_load_async_to_lds_b128 %0, %1, off"
                         :: "v"(lds), "v"(ga) : "memory");
        }
        // stage V chunk transposed: Vt[d][key] (VALU path; it's a transpose)
#pragma unroll
        for (int i = 0; i < 16; ++i) {
            int e = t + i * 256;                 // 0..4095
            int key = e >> 6, d = e & 63;
            Vt[d * LDV + key] = vbase[(size_t)key * ATT_D + d];
        }
        // prefetch next chunk into L2 while WMMAs run on this one
        if (kc + ATT_KC < ATT_S) {
            const __bf16* knext = kbase + (size_t)ATT_KC * ATT_D;
            const __bf16* vnext = vbase + (size_t)ATT_KC * ATT_D;
            int pk2 = t >> 2, pd = (t & 3) * 16;
            __builtin_prefetch(knext + (size_t)pk2 * ATT_D + pd, 0, 1);
            __builtin_prefetch(vnext + (size_t)pk2 * ATT_D + pd, 0, 1);
        }
        asm volatile("s_wait_asynccnt 0x0" ::: "memory");
        __syncthreads();

        // scores: four 16x16 tiles (key groups c*16)
        v8f s[4];
#pragma unroll
        for (int c = 0; c < 4; ++c)
#pragma unroll
            for (int r = 0; r < 8; ++r) s[c][r] = 0.0f;
#pragma unroll
        for (int f = 0; f < 2; ++f)
#pragma unroll
            for (int c = 0; c < 4; ++c) {
                Frag16 kb;
                load_b_frag(&Kt[(c * 16) * LDK + f * 32], LDK, lm, lh, kb);
                s[c] = wmma_bf16(qa[f], kb, s[c]);
            }

        // online softmax: row r+8*lh spans the 16 lanes of this half
        __bf16* Pw = &Ps[wave * 16 * LDP];
#pragma unroll
        for (int r = 0; r < 8; ++r) {
            float a0 = s[0][r], a1 = s[1][r], a2 = s[2][r], a3 = s[3][r];
            float tm = rowmax16(fmaxf(fmaxf(a0, a1), fmaxf(a2, a3)));
            float mnew = fmaxf(mrow[r], tm);
            float corr = __expf(mrow[r] - mnew);
            mrow[r] = mnew;
            float p0 = __expf(a0 - mnew);
            float p1 = __expf(a1 - mnew);
            float p2 = __expf(a2 - mnew);
            float p3 = __expf(a3 - mnew);
#pragma unroll
            for (int j = 0; j < 4; ++j) oacc[j][r] *= corr;
            osum[r] *= corr;
            int m = r + 8 * lh;
            Pw[m * LDP +  0 + lm] = (__bf16)p0;
            Pw[m * LDP + 16 + lm] = (__bf16)p1;
            Pw[m * LDP + 32 + lm] = (__bf16)p2;
            Pw[m * LDP + 48 + lm] = (__bf16)p3;
        }

        // O += P @ V ; osum += P @ 1  (per-wave LDS in-order; no barrier needed)
#pragma unroll
        for (int f = 0; f < 2; ++f) {
            Frag16 pa;
            load_a_frag(&Pw[f * 32], LDP, lm, lh, pa);
#pragma unroll
            for (int j = 0; j < 4; ++j) {
                Frag16 vb;
                load_b_frag(&Vt[(j * 16) * LDV + f * 32], LDV, lm, lh, vb);
                oacc[j] = wmma_bf16(pa, vb, oacc[j]);
            }
            osum = wmma_bf16(pa, onesf, osum);
        }
    }

    // normalize and store
#pragma unroll
    for (int j = 0; j < 4; ++j)
#pragma unroll
        for (int r = 0; r < 8; ++r) {
            int row = q0 + wave * 16 + r + 8 * lh;
            int col = (int)headoff + j * 16 + lm;
            float vv = oacc[j][r] / osum[r];
            o[((size_t)b * ATT_S + row) * ATT_D + col] = (__bf16)vv;
        }
}

// ---------------------------------------------------------------------------
extern "C" void kernel_launch(void* const* d_in, const int* in_sizes, int n_in,
                              void* d_out, int out_size, void* d_ws, size_t ws_size,
                              hipStream_t stream) {
    const float* x  = (const float*)d_in[0];
    const float* Wq = (const float*)d_in[1];
    const float* bq = (const float*)d_in[2];
    const float* Wk = (const float*)d_in[3];
    const float* bk = (const float*)d_in[4];
    const float* Wv = (const float*)d_in[5];
    const float* bv = (const float*)d_in[6];
    const float* Wo = (const float*)d_in[7];
    const float* bo = (const float*)d_in[8];
    float* out = (float*)d_out;

    const int M = 2 * 2048;       // B*S
    const int N = 1024, K = 1024;
    const size_t nEl = (size_t)M * N;

    __bf16* qb = (__bf16*)d_ws;
    __bf16* kb = qb + nEl;
    __bf16* vb = kb + nEl;
    __bf16* ab = vb + nEl;        // attention output (bf16)

    dim3 blk(256);
    dim3 ggrid(N / GT_N, M / GT_M);   // (8, 32)

    // fold 1/sqrt(Hd) = 0.125 into Q projection
    gemm_bias_kernel<false, true><<<ggrid, blk, 0, stream>>>(x, Wq, bq, qb, M, N, K, 0.125f);
    gemm_bias_kernel<false, true><<<ggrid, blk, 0, stream>>>(x, Wk, bk, kb, M, N, K, 1.0f);
    gemm_bias_kernel<false, true><<<ggrid, blk, 0, stream>>>(x, Wv, bv, vb, M, N, K, 1.0f);

    attn_kernel<<<dim3(2048 / 128, 16, 2), blk, 0, stream>>>(qb, kb, vb, ab);

    gemm_bias_kernel<true, false><<<ggrid, blk, 0, stream>>>(ab, Wo, bo, out, M, N, K, 1.0f);
}
